// DAGNN_10153302687992
// MI455X (gfx1250) — compile-verified, matrix operand order checked by reading, not run on hardware
//
#include <hip/hip_runtime.h>
#include <hip/hip_bf16.h>
#include <math.h>

// ---------------------------------------------------------------------------
// DAGNN for MI455X (gfx1250, wave32, WMMA).
//   MLP (256->128->128->16, f16 WMMA, f32 accum) -> pps[0]
//   10x edge scatter-add hops -> pps[1..10]
//   gated readout -> out (N x 16)
//
// MI455X notes:
//  - streaming data (x, edge_index, edge_attr, final pps sweep) uses
//    nontemporal loads so the 192MB L2 keeps the pps feature tables hot
//  - layer-2/3 packed weights staged to LDS via global_load_async_to_lds_b128
//    (ASYNCcnt), overlapped with layer-1 WMMA work
// ---------------------------------------------------------------------------

typedef _Float16 v16h __attribute__((ext_vector_type(16)));
typedef _Float16 v8h  __attribute__((ext_vector_type(8)));
typedef float    v8f  __attribute__((ext_vector_type(8)));
typedef float    v4f  __attribute__((ext_vector_type(4)));

#define NN     100000
#define EE     3200000
#define F_IN   256
#define HID    128
#define CC     16
#define KK     10

#define LROW   136            // padded LDS row stride in halves (bank-conflict free)
#define WPB    4              // waves per block (128 threads)
#define W23_BYTES  36864      // packed W2 (32KB) + W3 (4KB), contiguous in ws
#define W23_HALVES (W23_BYTES / 2)

// ---------------------------------------------------------------------------
// Pack weight matrix W (O x I, row-major fp32) into f16 WMMA-B fragments.
// B = W^T (I x O). Fragment tile (nt, kt): 32 lanes x 16 halves, contiguous.
// Lane l holds column n = nt*16 + (l&15); half = l>>4.
// Element e -> K = kt*32 + (e<8 ? half*8+e : 16 + half*8 + (e-8)).
// ---------------------------------------------------------------------------
__global__ void pack_weights(const float* __restrict__ W, int O, int I,
                             _Float16* __restrict__ out) {
    int t    = blockIdx.x * blockDim.x + threadIdx.x;
    int lane = t & 31;
    int tile = t >> 5;
    int kT   = I / 32;
    int nT   = O / 16;
    if (tile >= kT * nT) return;
    int nt   = tile / kT;
    int kt   = tile % kT;
    int half = lane >> 4;
    int n    = nt * 16 + (lane & 15);
    _Float16* o = out + ((size_t)tile * 32 + lane) * 16;
#pragma unroll
    for (int e = 0; e < 16; ++e) {
        int k = kt * 32 + ((e < 8) ? (half * 8 + e) : (16 + half * 8 + (e - 8)));
        o[e] = (_Float16)W[(size_t)n * I + k];
    }
}

// ---------------------------------------------------------------------------
// Fused 3-layer MLP. One wave handles a 16-node tile. 4 waves / block.
// ---------------------------------------------------------------------------
__global__ __launch_bounds__(128) void dagnn_mlp(
    const float* __restrict__ x,
    const float* __restrict__ b1, const float* __restrict__ b2,
    const float* __restrict__ b3,
    const _Float16* __restrict__ w1p, const _Float16* __restrict__ w23p,
    float* __restrict__ pps0) {

    __shared__ _Float16 lds[WPB][16 * LROW];                 // 17408 B
    __shared__ __align__(128) _Float16 wlds[W23_HALVES];     // 36864 B

    const int tid  = threadIdx.x;
    const int wave = tid >> 5;
    const int lane = tid & 31;
    const int m    = lane & 15;
    const int half = lane >> 4;
    const int nodeBase = (blockIdx.x * WPB + wave) * 16;

    int row = nodeBase + m;
    if (row >= NN) row = NN - 1;            // clamp; stores are guarded below

    // ---- kick off async staging of packed W2+W3 into LDS (ASYNCcnt) ----
    {
        typedef __attribute__((address_space(3))) char lds_char;
        unsigned lbase = (unsigned)(unsigned long long)(lds_char*)(void*)wlds;
        const char* gsrc = (const char*)w23p;
#pragma unroll
        for (int it = 0; it < W23_BYTES / (128 * 16); ++it) {   // 18 iters
            unsigned loff = lbase + (unsigned)(tid * 16 + it * 2048);
            unsigned long long ga = (unsigned long long)(gsrc + tid * 16 + it * 2048);
            asm volatile("global_load_async_to_lds_b128 %0, %1, off"
                         :: "v"(loff), "v"(ga) : "memory");
        }
    }

    // prefetch next block's row of x (global_prefetch_b8)
    {
        long pr = (long)row + WPB * 16;
        if (pr >= NN) pr = NN - 1;
        __builtin_prefetch(x + (size_t)pr * F_IN, 0, 3);
    }

    _Float16* L = lds[wave];

    // ---------------- Layer 1: (16x256) @ (256x128) ----------------
    v8f acc1[8] = {};
#pragma unroll
    for (int kt = 0; kt < F_IN / 32; ++kt) {
        // A fragment straight from global x (fp32 -> f16), nontemporal stream
        const float* pa = x + (size_t)row * F_IN + kt * 32 + half * 8;
        v4f f0 = __builtin_nontemporal_load((const v4f*)(pa));
        v4f f1 = __builtin_nontemporal_load((const v4f*)(pa + 4));
        v4f f2 = __builtin_nontemporal_load((const v4f*)(pa + 16));
        v4f f3 = __builtin_nontemporal_load((const v4f*)(pa + 20));
        v16h a;
#pragma unroll
        for (int i = 0; i < 4; ++i) {
            a[i]      = (_Float16)f0[i];
            a[i + 4]  = (_Float16)f1[i];
            a[i + 8]  = (_Float16)f2[i];
            a[i + 12] = (_Float16)f3[i];
        }
#pragma unroll
        for (int nt = 0; nt < HID / 16; ++nt) {
            v16h b = *(const v16h*)(w1p + ((size_t)(nt * (F_IN / 32) + kt) * 32 + lane) * 16);
            acc1[nt] = __builtin_amdgcn_wmma_f32_16x16x32_f16(
                false, a, false, b, (short)0, acc1[nt], false, false);
        }
    }

    // weights for layers 2/3 must have landed before anyone reads them
    asm volatile("s_wait_asynccnt 0x0" ::: "memory");
    __syncthreads();

    // bias + relu -> LDS (f16), C layout: VGPR r -> (row r+8*half, col lane&15)
#pragma unroll
    for (int nt = 0; nt < HID / 16; ++nt) {
        float bn = b1[nt * 16 + (lane & 15)];
#pragma unroll
        for (int r = 0; r < 8; ++r) {
            float v = acc1[nt][r] + bn;
            v = fmaxf(v, 0.0f);
            L[(r + 8 * half) * LROW + nt * 16 + (lane & 15)] = (_Float16)v;
        }
    }
    __syncthreads();

    const _Float16* w2l = wlds;               // 8*4 tiles
    const _Float16* w3l = wlds + 16384;       // 1*4 tiles

    // ---------------- Layer 2: (16x128) @ (128x128) ----------------
    v8f acc2[8] = {};
#pragma unroll
    for (int kt = 0; kt < HID / 32; ++kt) {
        const _Float16* lp = L + m * LROW + kt * 32 + half * 8;
        v8h lo = *(const v8h*)(lp);
        v8h hi = *(const v8h*)(lp + 16);
        v16h a;
#pragma unroll
        for (int i = 0; i < 8; ++i) { a[i] = lo[i]; a[i + 8] = hi[i]; }
#pragma unroll
        for (int nt = 0; nt < HID / 16; ++nt) {
            v16h b = *(const v16h*)(w2l + ((size_t)(nt * (HID / 32) + kt) * 32 + lane) * 16);
            acc2[nt] = __builtin_amdgcn_wmma_f32_16x16x32_f16(
                false, a, false, b, (short)0, acc2[nt], false, false);
        }
    }
    __syncthreads();
#pragma unroll
    for (int nt = 0; nt < HID / 16; ++nt) {
        float bn = b2[nt * 16 + (lane & 15)];
#pragma unroll
        for (int r = 0; r < 8; ++r) {
            float v = acc2[nt][r] + bn;
            v = fmaxf(v, 0.0f);
            L[(r + 8 * half) * LROW + nt * 16 + (lane & 15)] = (_Float16)v;
        }
    }
    __syncthreads();

    // ---------------- Layer 3: (16x128) @ (128x16) ----------------
    v8f acc3 = {};
#pragma unroll
    for (int kt = 0; kt < HID / 32; ++kt) {
        const _Float16* lp = L + m * LROW + kt * 32 + half * 8;
        v8h lo = *(const v8h*)(lp);
        v8h hi = *(const v8h*)(lp + 16);
        v16h a;
#pragma unroll
        for (int i = 0; i < 8; ++i) { a[i] = lo[i]; a[i + 8] = hi[i]; }
        v16h b = *(const v16h*)(w3l + ((size_t)kt * 32 + lane) * 16);
        acc3 = __builtin_amdgcn_wmma_f32_16x16x32_f16(
            false, a, false, b, (short)0, acc3, false, false);
    }
    // bias + relu -> pps[0]
    {
        float bn = b3[lane & 15];
#pragma unroll
        for (int r = 0; r < 8; ++r) {
            int mrow = nodeBase + r + 8 * half;
            if (mrow < NN) {
                float v = acc3[r] + bn;
                v = fmaxf(v, 0.0f);
                pps0[(size_t)mrow * CC + (lane & 15)] = v;
            }
        }
    }
}

// ---------------------------------------------------------------------------
// Zero a float buffer.
// ---------------------------------------------------------------------------
__global__ void zero_kernel(float* __restrict__ p, long n) {
    long i = (long)blockIdx.x * blockDim.x + threadIdx.x;
    if (i < n) p[i] = 0.0f;
}

// ---------------------------------------------------------------------------
// One propagation hop: next[dst] += attr * prev[src].  16 lanes per edge.
// Edge stream is read nontemporally (read-once, keep pps hot in L2).
// ---------------------------------------------------------------------------
__global__ __launch_bounds__(256) void hop_kernel(
    const int* __restrict__ ei, const float* __restrict__ ea,
    const float* __restrict__ prev, float* __restrict__ next) {
    long idx = (long)blockIdx.x * blockDim.x + threadIdx.x;
    long e = idx >> 4;
    int  c = (int)(idx & 15);
    if (e >= EE) return;
    int   s = __builtin_nontemporal_load(ei + e);
    int   d = __builtin_nontemporal_load(ei + (long)EE + e);
    float w = __builtin_nontemporal_load(ea + e);
    float v = w * prev[(size_t)s * CC + c];
    atomicAdd(&next[(size_t)d * CC + c], v);
}

// ---------------------------------------------------------------------------
// Readout: score = sigmoid(pps . pw + pb); out = sum_k score_k * pps_k
// 16 lanes per node; cross-lane reduce inside each 16-lane group (wave32).
// ---------------------------------------------------------------------------
__global__ __launch_bounds__(256) void readout_kernel(
    const float* __restrict__ pps, const float* __restrict__ proj_w,
    const float* __restrict__ proj_b, float* __restrict__ out) {
    long idx  = (long)blockIdx.x * blockDim.x + threadIdx.x;
    long node = idx >> 4;
    int  c    = (int)(idx & 15);
    if (node >= NN) return;
    float pw = proj_w[c];
    float pb = proj_b[0];
    float acc = 0.0f;
#pragma unroll
    for (int k = 0; k <= KK; ++k) {
        float v = __builtin_nontemporal_load(
            pps + (size_t)k * NN * CC + (size_t)node * CC + c);
        float p = v * pw;
        p += __shfl_xor(p, 1);
        p += __shfl_xor(p, 2);
        p += __shfl_xor(p, 4);
        p += __shfl_xor(p, 8);
        float s = 1.0f / (1.0f + expf(-(p + pb)));
        acc += s * v;
    }
    __builtin_nontemporal_store(acc, out + (size_t)node * CC + c);
}

// ---------------------------------------------------------------------------
extern "C" void kernel_launch(void* const* d_in, const int* in_sizes, int n_in,
                              void* d_out, int out_size, void* d_ws, size_t ws_size,
                              hipStream_t stream) {
    const float* x    = (const float*)d_in[0];
    const int*   ei   = (const int*)  d_in[1];   // (2, E) int32
    const float* ea   = (const float*)d_in[2];
    const float* w1   = (const float*)d_in[3];
    const float* b1   = (const float*)d_in[4];
    const float* w2   = (const float*)d_in[5];
    const float* b2   = (const float*)d_in[6];
    const float* w3   = (const float*)d_in[7];
    const float* b3   = (const float*)d_in[8];
    const float* pw   = (const float*)d_in[9];
    const float* pb   = (const float*)d_in[10];
    float* out = (float*)d_out;

    // workspace layout
    char* ws = (char*)d_ws;
    _Float16* w1p  = (_Float16*)(ws);                 // 8*8 tiles * 1024B = 64KB
    _Float16* w23p = (_Float16*)(ws + 65536);         // W2 (32KB) + W3 (4KB), contiguous
    _Float16* w2p  = w23p;
    _Float16* w3p  = (_Float16*)(ws + 65536 + 32768);
    float*    pps  = (float*)   (ws + 131072);        // (K+1)*N*16 f32

    // 1) pack weights into WMMA B-fragment layout (f16)
    pack_weights<<<(64 * 32 + 255) / 256, 256, 0, stream>>>(w1, HID, F_IN, w1p);
    pack_weights<<<(32 * 32 + 255) / 256, 256, 0, stream>>>(w2, HID, HID,  w2p);
    pack_weights<<<1, 128, 0, stream>>>(w3, CC, HID, w3p);

    // 2) zero hop accumulators pps[1..K]
    {
        long nz = (long)KK * NN * CC;
        zero_kernel<<<(unsigned)((nz + 255) / 256), 256, 0, stream>>>(pps + (size_t)NN * CC, nz);
    }

    // 3) fused MLP -> pps[0]
    dagnn_mlp<<<(NN + WPB * 16 - 1) / (WPB * 16), WPB * 32, 0, stream>>>(
        x, b1, b2, b3, w1p, w23p, pps);

    // 4) K propagation hops
    {
        long nthreads = (long)EE * CC;
        unsigned grid = (unsigned)((nthreads + 255) / 256);
        for (int k = 1; k <= KK; ++k) {
            hop_kernel<<<grid, 256, 0, stream>>>(
                ei, ea, pps + (size_t)(k - 1) * NN * CC, pps + (size_t)k * NN * CC);
        }
    }

    // 5) gated readout -> out
    {
        long nthreads = (long)NN * CC;
        readout_kernel<<<(unsigned)((nthreads + 255) / 256), 256, 0, stream>>>(
            pps, pw, pb, out);
    }
}
